// Attention_42442866819135
// MI455X (gfx1250) — matrix operand order repaired and checked by reading, not compiled, not run on hardware
//
#include <hip/hip_runtime.h>
#include <hip/hip_bf16.h>

typedef __bf16 bf16;
typedef __bf16 v16bf __attribute__((ext_vector_type(16)));
typedef float  v8f   __attribute__((ext_vector_type(8)));

#define WMMA_BF16(A, B, Cacc) \
  __builtin_amdgcn_wmma_f32_16x16x32_bf16(false, (A), false, (B), (short)0, (Cacc), false, false)

// Problem constants
#define BATCH 4
#define TLEN  2048
#define CDIM  512
#define HEADS 8
#define DHEAD 64
#define BT    (BATCH * TLEN)          // 8192
#define N_QKV (3 * CDIM)              // 1536

// Load a 16-element bf16 fragment from two 16-byte-aligned 8-element runs.
__device__ __forceinline__ v16bf load_frag2(const bf16* p0, const bf16* p1) {
  union { v16bf v; uint4 u[2]; } t;
  t.u[0] = *(const uint4*)p0;
  t.u[1] = *(const uint4*)p1;
  return t.v;
}

// ---------------------------------------------------------------------------
// Kernel 1: LayerNorm (fp32 in) -> bf16 out.  One 128-thread block per row.
// ---------------------------------------------------------------------------
__global__ __launch_bounds__(128) void ln_kernel(
    const float* __restrict__ x, const float* __restrict__ gamma,
    const float* __restrict__ beta, bf16* __restrict__ xn) {
  int row = blockIdx.x;
  int tid = threadIdx.x;
  const float* xr = x + (size_t)row * CDIM;
  float v[4], s = 0.f;
#pragma unroll
  for (int i = 0; i < 4; i++) { v[i] = xr[tid + i * 128]; s += v[i]; }
  __shared__ float red[128];
  red[tid] = s; __syncthreads();
  for (int off = 64; off > 0; off >>= 1) {
    if (tid < off) red[tid] += red[tid + off];
    __syncthreads();
  }
  float mu = red[0] * (1.0f / CDIM);
  __syncthreads();
  float s2 = 0.f;
#pragma unroll
  for (int i = 0; i < 4; i++) { float d = v[i] - mu; s2 += d * d; }
  red[tid] = s2; __syncthreads();
  for (int off = 64; off > 0; off >>= 1) {
    if (tid < off) red[tid] += red[tid + off];
    __syncthreads();
  }
  float rstd = rsqrtf(red[0] * (1.0f / CDIM) + 1e-5f);
#pragma unroll
  for (int i = 0; i < 4; i++) {
    int c = tid + i * 128;
    float y = (v[i] - mu) * rstd * gamma[c] + beta[c];
    xn[(size_t)row * CDIM + c] = (bf16)y;
  }
}

// ---------------------------------------------------------------------------
// Kernel: fp32 [K,N] -> bf16 transposed [N,K]  (weight prep)
// ---------------------------------------------------------------------------
__global__ void cvt_t_kernel(const float* __restrict__ src, bf16* __restrict__ dst,
                             int K, int N) {
  int i = blockIdx.x * 256 + threadIdx.x;
  if (i < K * N) {
    int n = i / K, kk = i % K;
    dst[i] = (bf16)src[(size_t)kk * N + n];
  }
}

// ---------------------------------------------------------------------------
// Kernel 2: QKV GEMM  xn[8192,512](bf16) @ wqkvT[1536,512](bf16, pre-transposed)
// One wave computes a 16x64 strip (4 N-tiles sharing the A fragment).
// Software-pipelined: fragments for k+1 loaded before the WMMAs of k.
// q,k scattered to [B,H,T,D]; v scattered TRANSPOSED to [B,H,D,T].
// ---------------------------------------------------------------------------
__global__ __launch_bounds__(256) void qkv_gemm(
    const bf16* __restrict__ A, const bf16* __restrict__ BT_,
    bf16* __restrict__ q, bf16* __restrict__ k, bf16* __restrict__ vT) {
  int wave = blockIdx.x * 8 + (threadIdx.x >> 5);
  int lane = threadIdx.x & 31;
  int lh = lane >> 4, lr = lane & 15;
  int mt = wave / (N_QKV / 64);         // 0..511
  int ng = wave % (N_QKV / 64);         // 0..23 (groups of 4 N-tiles)
  const bf16* arow = A + (size_t)(mt * 16 + lr) * CDIM;
  const bf16* brow[4];
  int ncol[4];
#pragma unroll
  for (int i = 0; i < 4; i++) {
    ncol[i] = ng * 64 + i * 16 + lr;
    brow[i] = BT_ + (size_t)ncol[i] * CDIM;
  }
  v8f acc0 = {}, acc1 = {}, acc2 = {}, acc3 = {};
  // prologue loads (k0 = 0)
  v16bf ac  = load_frag2(arow + lh * 16, arow + lh * 16 + 8);
  v16bf b0c = load_frag2(brow[0] + lh * 16, brow[0] + lh * 16 + 8);
  v16bf b1c = load_frag2(brow[1] + lh * 16, brow[1] + lh * 16 + 8);
  v16bf b2c = load_frag2(brow[2] + lh * 16, brow[2] + lh * 16 + 8);
  v16bf b3c = load_frag2(brow[3] + lh * 16, brow[3] + lh * 16 + 8);
#pragma unroll
  for (int k0 = 0; k0 < CDIM; k0 += 32) {
    int kn = k0 + 32;
    v16bf an = ac, b0n = b0c, b1n = b1c, b2n = b2c, b3n = b3c;
    if (kn < CDIM) {
      an  = load_frag2(arow + kn + lh * 16, arow + kn + lh * 16 + 8);
      b0n = load_frag2(brow[0] + kn + lh * 16, brow[0] + kn + lh * 16 + 8);
      b1n = load_frag2(brow[1] + kn + lh * 16, brow[1] + kn + lh * 16 + 8);
      b2n = load_frag2(brow[2] + kn + lh * 16, brow[2] + kn + lh * 16 + 8);
      b3n = load_frag2(brow[3] + kn + lh * 16, brow[3] + kn + lh * 16 + 8);
    }
    acc0 = WMMA_BF16(ac, b0c, acc0);
    acc1 = WMMA_BF16(ac, b1c, acc1);
    acc2 = WMMA_BF16(ac, b2c, acc2);
    acc3 = WMMA_BF16(ac, b3c, acc3);
    ac = an; b0c = b0n; b1c = b1n; b2c = b2n; b3c = b3n;
  }
  v8f accs[4] = {acc0, acc1, acc2, acc3};
#pragma unroll
  for (int i = 0; i < 4; i++) {
    int which = ncol[i] / CDIM;          // uniform per wave (64-wide group)
    int inner = ncol[i] % CDIM;
    int h = inner >> 6, d = inner & 63;
#pragma unroll
    for (int r = 0; r < 8; r++) {
      int row = mt * 16 + lh * 8 + r;
      int bi = row >> 11, t = row & 2047;
      size_t bh = (size_t)bi * HEADS + h;
      if (which == 2) {
        vT[(bh * DHEAD + d) * TLEN + t] = (bf16)accs[i][r];
      } else {
        bf16* dst = (which == 0) ? q : k;
        dst[(bh * TLEN + t) * DHEAD + d] = (bf16)accs[i][r];
      }
    }
  }
}

// ---------------------------------------------------------------------------
// Kernel 3: block-causal flash attention. One wave per (b,h,16-row q tile).
// Mask is tile-aligned (frames of 256) -> loop over (frame+1)*16 key tiles.
// V is stored transposed [B,H,D,T]; all K/V fragment loads issue at the top
// of the iteration so they overlap S-WMMAs + softmax TRANS ops.
// ---------------------------------------------------------------------------
__global__ __launch_bounds__(256) void attn_kernel(
    const bf16* __restrict__ q, const bf16* __restrict__ k,
    const bf16* __restrict__ vT, bf16* __restrict__ o) {
  __shared__ __align__(16) bf16 pshare[8][16 * 32];   // per-wave P scratch
  int wslot = threadIdx.x >> 5;
  int wave = blockIdx.x * 8 + wslot;
  int lane = threadIdx.x & 31;
  int lh = lane >> 4, lr = lane & 15;
  int bh = wave >> 7;                   // 0..31
  int qt = wave & 127;                  // 0..127
  const bf16* qb = q + (size_t)bh * TLEN * DHEAD;
  const bf16* kb = k + (size_t)bh * TLEN * DHEAD;
  const bf16* vb = vT + (size_t)bh * DHEAD * TLEN;

  // Q fragments (16x64 -> two K=32 A-frags), kept resident
  const bf16* qrow = qb + (size_t)(qt * 16 + lr) * DHEAD;
  v16bf aq0 = load_frag2(qrow + lh * 16, qrow + lh * 16 + 8);
  v16bf aq1 = load_frag2(qrow + 32 + lh * 16, qrow + 40 + lh * 16);

  v8f o0 = {}, o1 = {}, o2 = {}, o3 = {};
  float mrow[8], lrow[8];
#pragma unroll
  for (int r = 0; r < 8; r++) { mrow[r] = -1e30f; lrow[r] = 0.f; }

  const float scale = 0.125f;           // 1/sqrt(64)
  int npairs = ((qt >> 4) + 1) * 8;     // (frame+1)*16 key tiles / 2
  bf16* pl = pshare[wslot];

  for (int p = 0; p < npairs; p++) {
    int keybase = p * 32;
    // --- issue ALL K and V fragment loads for this pair up front ---
    const bf16* krow0 = kb + (size_t)(keybase + lr) * DHEAD + lh * 16;
    const bf16* krow1 = kb + (size_t)(keybase + 16 + lr) * DHEAD + lh * 16;
    v16bf bk00 = load_frag2(krow0, krow0 + 8);
    v16bf bk01 = load_frag2(krow0 + 32, krow0 + 40);
    v16bf bk10 = load_frag2(krow1, krow1 + 8);
    v16bf bk11 = load_frag2(krow1 + 32, krow1 + 40);
    const bf16* vp0 = vb + (size_t)(0  + lr) * TLEN + keybase + lh * 16;
    const bf16* vp1 = vb + (size_t)(16 + lr) * TLEN + keybase + lh * 16;
    const bf16* vp2 = vb + (size_t)(32 + lr) * TLEN + keybase + lh * 16;
    const bf16* vp3 = vb + (size_t)(48 + lr) * TLEN + keybase + lh * 16;
    v16bf bv0 = load_frag2(vp0, vp0 + 8);
    v16bf bv1 = load_frag2(vp1, vp1 + 8);
    v16bf bv2 = load_frag2(vp2, vp2 + 8);
    v16bf bv3 = load_frag2(vp3, vp3 + 8);
    // prefetch next key-pair's K/V cachelines (global_prefetch_b8)
    if (p + 1 < npairs) {
      __builtin_prefetch(kb + (size_t)(keybase + 32 + lr) * DHEAD, 0, 3);
      __builtin_prefetch(vb + (size_t)lr * TLEN + keybase + 32, 0, 3);
    }
    // --- S = Q @ K^T for two 16-key subtiles ---
    v8f s0 = {}, s1 = {};
    s0 = WMMA_BF16(aq0, bk00, s0);
    s0 = WMMA_BF16(aq1, bk01, s0);
    s1 = WMMA_BF16(aq0, bk10, s1);
    s1 = WMMA_BF16(aq1, bk11, s1);
    // --- online softmax (rows live at fixed lane-half; reduce over 16 lanes) ---
    float alpha[8];
#pragma unroll
    for (int r = 0; r < 8; r++) {
      float x0 = s0[r] * scale, x1 = s1[r] * scale;
      float mx = fmaxf(x0, x1);
      mx = fmaxf(mx, __shfl_xor(mx, 1, 32));
      mx = fmaxf(mx, __shfl_xor(mx, 2, 32));
      mx = fmaxf(mx, __shfl_xor(mx, 4, 32));
      mx = fmaxf(mx, __shfl_xor(mx, 8, 32));
      float mnew = fmaxf(mrow[r], mx);
      float p0v = __expf(x0 - mnew);
      float p1v = __expf(x1 - mnew);
      float sum = p0v + p1v;
      sum += __shfl_xor(sum, 1, 32);
      sum += __shfl_xor(sum, 2, 32);
      sum += __shfl_xor(sum, 4, 32);
      sum += __shfl_xor(sum, 8, 32);
      alpha[r] = __expf(mrow[r] - mnew);
      lrow[r] = lrow[r] * alpha[r] + sum;
      mrow[r] = mnew;
      // C-layout -> LDS (row = lh*8+r, cols lr and 16+lr)
      pl[(lh * 8 + r) * 32 + lr]      = (bf16)p0v;
      pl[(lh * 8 + r) * 32 + 16 + lr] = (bf16)p1v;
    }
    // --- read P back in A-layout (LDS ops are in-order within a wave) ---
    v16bf pa = load_frag2(&pl[lr * 32 + lh * 16], &pl[lr * 32 + lh * 16 + 8]);
    // rescale running O by alpha before accumulation
#pragma unroll
    for (int r = 0; r < 8; r++) {
      o0[r] *= alpha[r]; o1[r] *= alpha[r];
      o2[r] *= alpha[r]; o3[r] *= alpha[r];
    }
    o0 = WMMA_BF16(pa, bv0, o0);
    o1 = WMMA_BF16(pa, bv1, o1);
    o2 = WMMA_BF16(pa, bv2, o2);
    o3 = WMMA_BF16(pa, bv3, o3);
  }

  // epilogue: normalize and write [B,T,H*D] bf16
  int b_ = bh >> 3, h_ = bh & 7;
#pragma unroll
  for (int r = 0; r < 8; r++) {
    float inv = 1.0f / lrow[r];
    size_t rowoff = ((size_t)b_ * TLEN + qt * 16 + lh * 8 + r) * CDIM + h_ * DHEAD + lr;
    o[rowoff]      = (bf16)(o0[r] * inv);
    o[rowoff + 16] = (bf16)(o1[r] * inv);
    o[rowoff + 32] = (bf16)(o2[r] * inv);
    o[rowoff + 48] = (bf16)(o3[r] * inv);
  }
}

// ---------------------------------------------------------------------------
// Kernel 4: attn[8192,512] @ w_outT[512,512](pre-transposed) + b_out -> fp32
// One wave computes a 16x64 strip (4 N-tiles), software-pipelined K loop.
// ---------------------------------------------------------------------------
__global__ __launch_bounds__(256) void out_gemm(
    const bf16* __restrict__ A, const bf16* __restrict__ BT_,
    const float* __restrict__ bias, float* __restrict__ out) {
  int wave = blockIdx.x * 8 + (threadIdx.x >> 5);
  int lane = threadIdx.x & 31;
  int lh = lane >> 4, lr = lane & 15;
  int mt = wave >> 3;                   // 512 M tiles
  int ng = wave & 7;                    // 8 groups of 4 N-tiles
  const bf16* arow = A + (size_t)(mt * 16 + lr) * CDIM;
  const bf16* brow[4];
  int ncol[4];
#pragma unroll
  for (int i = 0; i < 4; i++) {
    ncol[i] = ng * 64 + i * 16 + lr;
    brow[i] = BT_ + (size_t)ncol[i] * CDIM;
  }
  v8f acc0 = {}, acc1 = {}, acc2 = {}, acc3 = {};
  v16bf ac  = load_frag2(arow + lh * 16, arow + lh * 16 + 8);
  v16bf b0c = load_frag2(brow[0] + lh * 16, brow[0] + lh * 16 + 8);
  v16bf b1c = load_frag2(brow[1] + lh * 16, brow[1] + lh * 16 + 8);
  v16bf b2c = load_frag2(brow[2] + lh * 16, brow[2] + lh * 16 + 8);
  v16bf b3c = load_frag2(brow[3] + lh * 16, brow[3] + lh * 16 + 8);
#pragma unroll
  for (int k0 = 0; k0 < CDIM; k0 += 32) {
    int kn = k0 + 32;
    v16bf an = ac, b0n = b0c, b1n = b1c, b2n = b2c, b3n = b3c;
    if (kn < CDIM) {
      an  = load_frag2(arow + kn + lh * 16, arow + kn + lh * 16 + 8);
      b0n = load_frag2(brow[0] + kn + lh * 16, brow[0] + kn + lh * 16 + 8);
      b1n = load_frag2(brow[1] + kn + lh * 16, brow[1] + kn + lh * 16 + 8);
      b2n = load_frag2(brow[2] + kn + lh * 16, brow[2] + kn + lh * 16 + 8);
      b3n = load_frag2(brow[3] + kn + lh * 16, brow[3] + kn + lh * 16 + 8);
    }
    acc0 = WMMA_BF16(ac, b0c, acc0);
    acc1 = WMMA_BF16(ac, b1c, acc1);
    acc2 = WMMA_BF16(ac, b2c, acc2);
    acc3 = WMMA_BF16(ac, b3c, acc3);
    ac = an; b0c = b0n; b1c = b1n; b2c = b2n; b3c = b3n;
  }
  v8f accs[4] = {acc0, acc1, acc2, acc3};
#pragma unroll
  for (int i = 0; i < 4; i++) {
    float bv = bias[ncol[i]];
#pragma unroll
    for (int r = 0; r < 8; r++) {
      out[(size_t)(mt * 16 + lh * 8 + r) * CDIM + ncol[i]] = accs[i][r] + bv;
    }
  }
}

// ---------------------------------------------------------------------------
extern "C" void kernel_launch(void* const* d_in, const int* in_sizes, int n_in,
                              void* d_out, int out_size, void* d_ws, size_t ws_size,
                              hipStream_t stream) {
  const float* x     = (const float*)d_in[0];
  const float* gam   = (const float*)d_in[1];
  const float* bet   = (const float*)d_in[2];
  const float* wqkv  = (const float*)d_in[3];
  const float* wout  = (const float*)d_in[4];
  const float* bout  = (const float*)d_in[5];
  float* out = (float*)d_out;

  bf16* ws      = (bf16*)d_ws;
  bf16* xn      = ws;                                   // [8192,512]; reused as attn out
  bf16* wqkvT   = xn + (size_t)BT * CDIM;               // [1536,512]
  bf16* woutT   = wqkvT + (size_t)CDIM * N_QKV;         // [512,512]
  bf16* qws     = woutT + (size_t)CDIM * CDIM;          // [B,H,T,D]
  bf16* kws     = qws + (size_t)BATCH * HEADS * TLEN * DHEAD;
  bf16* vwsT    = kws + (size_t)BATCH * HEADS * TLEN * DHEAD;  // [B,H,D,T]

  ln_kernel<<<BT, 128, 0, stream>>>(x, gam, bet, xn);
  cvt_t_kernel<<<(CDIM * N_QKV + 255) / 256, 256, 0, stream>>>(wqkv, wqkvT, CDIM, N_QKV);
  cvt_t_kernel<<<(CDIM * CDIM + 255) / 256, 256, 0, stream>>>(wout, woutT, CDIM, CDIM);
  // 8192/16 * 1536/64 = 12288 waves -> 1536 blocks of 8 waves
  qkv_gemm<<<1536, 256, 0, stream>>>(xn, wqkvT, qws, kws, vwsT);
  // 4*8*128 = 4096 waves -> 512 blocks
  attn_kernel<<<512, 256, 0, stream>>>(qws, kws, vwsT, xn);
  // 512*8 = 4096 waves -> 512 blocks
  out_gemm<<<512, 256, 0, stream>>>(xn, woutT, bout, out);
}
